// TreeCell_13864154432276
// MI455X (gfx1250) — matrix operand order repaired and checked by reading, not compiled, not run on hardware
//
#include <hip/hip_runtime.h>
#include <hip/hip_bf16.h>
#include <math.h>

// Perfect binary tree TreeCell on gfx1250 (MI455X).
// All GEMMs run on v_wmma_f32_16x16x32_f16 with split-f16 (hi/lo) inputs for
// fp32-class accuracy at f16 WMMA issue rate. Weights staged in LDS pre-split.

#define DEPTH 17
#define NNODES ((1 << (DEPTH + 1)) - 1)   // 262143
#define SZ 128
#define LEAF_START ((1 << DEPTH) - 1)     // 131071
#define TOPL 10                           // levels 0..TOPL handled by 1-block kernels

typedef __attribute__((ext_vector_type(16))) _Float16 v16h;
typedef __attribute__((ext_vector_type(8)))  float    v8f;
typedef __attribute__((ext_vector_type(4)))  float    f32x4;
typedef __attribute__((ext_vector_type(4)))  unsigned int u32x4;

union V16H { v16h v; _Float16 h[16]; u32x4 q[2]; };

// ---------------------------------------------------------------- helpers ---

// Split 16 fp32 values (4x float4) into hi/lo f16 fragments.
__device__ __forceinline__ void split_frag(f32x4 a0, f32x4 a1, f32x4 a2, f32x4 a3,
                                           v16h& hi, v16h& lo) {
  V16H H, L;
#pragma unroll
  for (int i = 0; i < 4; ++i) {
    float x; _Float16 h;
    x = a0[i]; h = (_Float16)x; H.h[i]      = h; L.h[i]      = (_Float16)(x - (float)h);
    x = a1[i]; h = (_Float16)x; H.h[4 + i]  = h; L.h[4 + i]  = (_Float16)(x - (float)h);
    x = a2[i]; h = (_Float16)x; H.h[8 + i]  = h; L.h[8 + i]  = (_Float16)(x - (float)h);
    x = a3[i]; h = (_Float16)x; H.h[12 + i] = h; L.h[12 + i] = (_Float16)(x - (float)h);
  }
  hi = H.v; lo = L.v;
}

// A fragment (16x32 f16, ISA layout: lane holds row M=lane&15; halves 0..7 ->
// K = 8*(lane>>4)+i, halves 8..15 -> K = 16+8*(lane>>4)+(i-8)), from one row.
__device__ __forceinline__ void a_frag_1(const float* __restrict__ row, int ks, int lane,
                                         v16h& hi, v16h& lo) {
  int k0 = 32 * ks + ((lane >> 4) << 3);
  f32x4 a0 = *(const f32x4*)(row + k0);
  f32x4 a1 = *(const f32x4*)(row + k0 + 4);
  f32x4 a2 = *(const f32x4*)(row + k0 + 16);
  f32x4 a3 = *(const f32x4*)(row + k0 + 20);
  split_frag(a0, a1, a2, a3, hi, lo);
}

// Same, but A row = sum of two child rows (fp32 add before split).
__device__ __forceinline__ void a_frag_2(const float* __restrict__ r0,
                                         const float* __restrict__ r1, int ks, int lane,
                                         v16h& hi, v16h& lo) {
  int k0 = 32 * ks + ((lane >> 4) << 3);
  f32x4 a0 = *(const f32x4*)(r0 + k0)      + *(const f32x4*)(r1 + k0);
  f32x4 a1 = *(const f32x4*)(r0 + k0 + 4)  + *(const f32x4*)(r1 + k0 + 4);
  f32x4 a2 = *(const f32x4*)(r0 + k0 + 16) + *(const f32x4*)(r1 + k0 + 16);
  f32x4 a3 = *(const f32x4*)(r0 + k0 + 20) + *(const f32x4*)(r1 + k0 + 20);
  split_frag(a0, a1, a2, a3, hi, lo);
}

// B fragment (32x16 f16). ISA wave32 B layout: lane holds column n=lane&15,
// halves 0..7 -> K = 8*(lane>>4)+i, halves 8..15 -> K = 16+8*(lane>>4)+(i-8).
// LDS weight array is W[j][k] row-major (B[k][j] = W[j][k]).
__device__ __forceinline__ v16h b_frag(const _Float16* __restrict__ Wl,
                                       int ctile, int ks, int lane) {
  const _Float16* p = Wl + (ctile * 16 + (lane & 15)) * SZ + ks * 32 + ((lane >> 4) << 3);
  V16H u;
  u.q[0] = *(const u32x4*)(p);
  u.q[1] = *(const u32x4*)(p + 16);
  return u.v;
}

// 16-row x 128-col strip GEMM: acc += A(16x128) @ W^T using split-f16
// (3 products per k-step) -> 96 WMMAs per strip.
__device__ __forceinline__ void gemm8(const v16h ahi[4], const v16h alo[4],
                                      const _Float16* __restrict__ sWhi,
                                      const _Float16* __restrict__ sWlo,
                                      int lane, v8f acc[8]) {
#pragma unroll
  for (int c = 0; c < 8; ++c) {
    v8f C = acc[c];
#pragma unroll
    for (int ks = 0; ks < 4; ++ks) {
      v16h bh = b_frag(sWhi, c, ks, lane);
      v16h bl = b_frag(sWlo, c, ks, lane);
      C = __builtin_amdgcn_wmma_f32_16x16x32_f16(false, ahi[ks], false, bh, (short)0, C, false, false);
      C = __builtin_amdgcn_wmma_f32_16x16x32_f16(false, alo[ks], false, bh, (short)0, C, false, false);
      C = __builtin_amdgcn_wmma_f32_16x16x32_f16(false, ahi[ks], false, bl, (short)0, C, false, false);
    }
    acc[c] = C;
  }
}

// Stage a 128x128 fp32 weight matrix into LDS as split-f16 hi/lo.
__device__ __forceinline__ void stage_w(const float* __restrict__ W,
                                        _Float16* hi, _Float16* lo) {
  for (int i = threadIdx.x; i < SZ * SZ; i += 256) {
    float w = W[i];
    _Float16 h = (_Float16)w;
    hi[i] = h;
    lo[i] = (_Float16)(w - (float)h);
  }
}

// ----------------------------------------------------------- strip bodies ---

// Inside pass, one 16-node strip at level start s (cnt nodes), offset q0.
// slot0: holds proj_in for this level / inside_h for children; slot1: inside_ex.
__device__ __forceinline__ void inside_strip(float* __restrict__ slot0,
                                             float* __restrict__ slot1,
                                             const _Float16* __restrict__ sWhi,
                                             const _Float16* __restrict__ sWlo,
                                             const float* __restrict__ bias,
                                             int s, int cnt, int q0, int lane) {
  int node = s + q0 + (lane & 15);
  const float* c0 = slot0 + (size_t)(2 * node + 1) * SZ;
  const float* c1 = slot0 + (size_t)(2 * node + 2) * SZ;
  v16h ahi[4], alo[4];
#pragma unroll
  for (int ks = 0; ks < 4; ++ks) a_frag_2(c0, c1, ks, lane, ahi[ks], alo[ks]);
  v8f acc[8];
#pragma unroll
  for (int c = 0; c < 8; ++c) {
    float bv = 2.0f * bias[c * 16 + (lane & 15)];
#pragma unroll
    for (int r = 0; r < 8; ++r) acc[c][r] = bv;
  }
  gemm8(ahi, alo, sWhi, sWlo, lane, acc);
  int rbase = (lane >> 4) << 3;
#pragma unroll
  for (int c = 0; c < 8; ++c) {
    int col = c * 16 + (lane & 15);
#pragma unroll
    for (int r = 0; r < 8; ++r) {
      int q = q0 + rbase + r;
      if (q < cnt) {
        size_t off = (size_t)(s + q) * SZ + col;
        float red = acc[c][r];
        slot1[off] = red;                         // inside_ex
        slot0[off] = tanhf(slot0[off] + red);     // proj_in -> inside_h
      }
    }
  }
}

// Outside pass, one 16-node strip. slot2: proj_out->outside_h; slot3: outside_ex.
__device__ __forceinline__ void outside_strip(float* __restrict__ slot2,
                                              float* __restrict__ slot3,
                                              const float* __restrict__ slot0,
                                              const _Float16* __restrict__ sWhi,
                                              const _Float16* __restrict__ sWlo,
                                              const float* __restrict__ bias,
                                              int s, int cnt, int ps, int q0, int lane) {
  int qa = q0 + (lane & 15);
  const float* prow = slot2 + (size_t)(ps + (qa >> 1)) * SZ;   // parent outside_h
  v16h ahi[4], alo[4];
#pragma unroll
  for (int ks = 0; ks < 4; ++ks) a_frag_1(prow, ks, lane, ahi[ks], alo[ks]);
  v8f acc[8];
  int rbase = (lane >> 4) << 3;
#pragma unroll
  for (int c = 0; c < 8; ++c) {
    int col = c * 16 + (lane & 15);
    float bv = bias[col];
#pragma unroll
    for (int r = 0; r < 8; ++r) {
      int q = q0 + rbase + r;
      acc[c][r] = bv + slot0[(size_t)(s + (q ^ 1)) * SZ + col]; // + sibling inside_h
    }
  }
  gemm8(ahi, alo, sWhi, sWlo, lane, acc);
#pragma unroll
  for (int c = 0; c < 8; ++c) {
    int col = c * 16 + (lane & 15);
#pragma unroll
    for (int r = 0; r < 8; ++r) {
      int q = q0 + rbase + r;
      if (q < cnt) {
        size_t off = (size_t)(s + q) * SZ + col;
        float red = acc[c][r];
        slot3[off] = red;                         // outside_ex
        slot2[off] = tanhf(slot2[off] + red);     // proj_out -> outside_h
      }
    }
  }
}

// ---------------------------------------------------------------- kernels ---

// proj = feat @ W^T for all rows; leaves (row >= leaf_start): dst=tanh, ex=0.
__global__ void __launch_bounds__(256)
proj_kernel(const float* __restrict__ feat, const float* __restrict__ W,
            float* __restrict__ dst, float* __restrict__ exz,
            int nrows, int leaf_start) {
  __shared__ _Float16 sWhi[SZ * SZ];
  __shared__ _Float16 sWlo[SZ * SZ];
  stage_w(W, sWhi, sWlo);
  __syncthreads();
  int lane = threadIdx.x & 31, wave = threadIdx.x >> 5;
  int row0 = (blockIdx.x * 8 + wave) * 16;
  if (row0 >= nrows) return;
  int arow = row0 + (lane & 15);
  if (arow > nrows - 1) arow = nrows - 1;       // clamp for partial last strip
  const float* rowp = feat + (size_t)arow * SZ;
  v16h ahi[4], alo[4];
#pragma unroll
  for (int ks = 0; ks < 4; ++ks) a_frag_1(rowp, ks, lane, ahi[ks], alo[ks]);
  v8f acc[8];
#pragma unroll
  for (int c = 0; c < 8; ++c)
#pragma unroll
    for (int r = 0; r < 8; ++r) acc[c][r] = 0.0f;
  gemm8(ahi, alo, sWhi, sWlo, lane, acc);
  int rbase = (lane >> 4) << 3;
#pragma unroll
  for (int c = 0; c < 8; ++c) {
    int col = c * 16 + (lane & 15);
#pragma unroll
    for (int r = 0; r < 8; ++r) {
      int row = row0 + rbase + r;
      if (row < nrows) {
        size_t off = (size_t)row * SZ + col;
        float v = acc[c][r];
        if (row >= leaf_start) { dst[off] = tanhf(v); exz[off] = 0.0f; }
        else                   { dst[off] = v; }
      }
    }
  }
}

__global__ void __launch_bounds__(256)
inside_level_kernel(float* __restrict__ slot0, float* __restrict__ slot1,
                    const float* __restrict__ W, const float* __restrict__ bias,
                    int s, int cnt) {
  __shared__ _Float16 sWhi[SZ * SZ];
  __shared__ _Float16 sWlo[SZ * SZ];
  stage_w(W, sWhi, sWlo);
  __syncthreads();
  int lane = threadIdx.x & 31, wave = threadIdx.x >> 5;
  int q0 = (blockIdx.x * 8 + wave) * 16;
  inside_strip(slot0, slot1, sWhi, sWlo, bias, s, cnt, q0, lane);
}

__global__ void __launch_bounds__(256)
outside_level_kernel(float* __restrict__ slot2, float* __restrict__ slot3,
                     const float* __restrict__ slot0,
                     const float* __restrict__ W, const float* __restrict__ bias,
                     int s, int cnt, int ps) {
  __shared__ _Float16 sWhi[SZ * SZ];
  __shared__ _Float16 sWlo[SZ * SZ];
  stage_w(W, sWhi, sWlo);
  __syncthreads();
  int lane = threadIdx.x & 31, wave = threadIdx.x >> 5;
  int q0 = (blockIdx.x * 8 + wave) * 16;
  outside_strip(slot2, slot3, slot0, sWhi, sWlo, bias, s, cnt, ps, q0, lane);
}

// Single block walks inside levels TOPL..0 with intra-block barriers.
__global__ void __launch_bounds__(256)
inside_top_kernel(float* __restrict__ slot0, float* __restrict__ slot1,
                  const float* __restrict__ W, const float* __restrict__ bias) {
  __shared__ _Float16 sWhi[SZ * SZ];
  __shared__ _Float16 sWlo[SZ * SZ];
  stage_w(W, sWhi, sWlo);
  __syncthreads();
  int lane = threadIdx.x & 31, wave = threadIdx.x >> 5;
  for (int l = TOPL; l >= 0; --l) {
    int cnt = 1 << l, s = cnt - 1;
    int strips = (cnt + 15) >> 4;
    for (int strip = wave; strip < strips; strip += 8)
      inside_strip(slot0, slot1, sWhi, sWlo, bias, s, cnt, strip * 16, lane);
    __syncthreads();
  }
}

// Single block: root init + outside levels 1..TOPL.
__global__ void __launch_bounds__(256)
outside_top_kernel(float* __restrict__ slot2, float* __restrict__ slot3,
                   const float* __restrict__ slot0,
                   const float* __restrict__ W, const float* __restrict__ bias) {
  __shared__ _Float16 sWhi[SZ * SZ];
  __shared__ _Float16 sWlo[SZ * SZ];
  stage_w(W, sWhi, sWlo);
  if (threadIdx.x < SZ) {                       // root: outside_h = tanh(proj_out), ex = 0
    slot3[threadIdx.x] = 0.0f;
    slot2[threadIdx.x] = tanhf(slot2[threadIdx.x]);
  }
  __syncthreads();
  int lane = threadIdx.x & 31, wave = threadIdx.x >> 5;
  for (int l = 1; l <= TOPL; ++l) {
    int cnt = 1 << l, s = cnt - 1, ps = (cnt >> 1) - 1;
    int strips = (cnt + 15) >> 4;
    for (int strip = wave; strip < strips; strip += 8)
      outside_strip(slot2, slot3, slot0, sWhi, sWlo, bias, s, cnt, ps, strip * 16, lane);
    __syncthreads();
  }
}

// ------------------------------------------------------------------ launch ---

extern "C" void kernel_launch(void* const* d_in, const int* in_sizes, int n_in,
                              void* d_out, int out_size, void* d_ws, size_t ws_size,
                              hipStream_t stream) {
  (void)in_sizes; (void)n_in; (void)out_size; (void)d_ws; (void)ws_size;
  const float* feat       = (const float*)d_in[0];
  const float* W_in_node  = (const float*)d_in[1];
  const float* W_in_inc   = (const float*)d_in[2];
  const float* b_in_inc   = (const float*)d_in[3];
  const float* W_out_node = (const float*)d_in[4];
  const float* W_out_inc  = (const float*)d_in[5];
  const float* b_out_inc  = (const float*)d_in[6];
  float* out = (float*)d_out;
  size_t slotsz = (size_t)NNODES * SZ;
  float* s0 = out;               // inside_h   (initially proj_in)
  float* s1 = out + slotsz;      // inside_ex
  float* s2 = out + 2 * slotsz;  // outside_h  (initially proj_out)
  float* s3 = out + 3 * slotsz;  // outside_ex
  dim3 blk(256);

  int strips = (NNODES + 15) / 16;           // 16384
  int pblocks = (strips + 7) / 8;            // 2048
  proj_kernel<<<pblocks, blk, 0, stream>>>(feat, W_in_node,  s0, s1, NNODES, LEAF_START);
  proj_kernel<<<pblocks, blk, 0, stream>>>(feat, W_out_node, s2, s3, NNODES, NNODES);

  for (int l = DEPTH - 1; l > TOPL; --l) {   // inside: big levels 16..11
    int cnt = 1 << l, s = cnt - 1;
    inside_level_kernel<<<cnt / 128, blk, 0, stream>>>(s0, s1, W_in_inc, b_in_inc, s, cnt);
  }
  inside_top_kernel<<<1, blk, 0, stream>>>(s0, s1, W_in_inc, b_in_inc);    // levels 10..0

  outside_top_kernel<<<1, blk, 0, stream>>>(s2, s3, s0, W_out_inc, b_out_inc); // root + 1..10
  for (int l = TOPL + 1; l <= DEPTH; ++l) {  // outside: big levels 11..17
    int cnt = 1 << l, s = cnt - 1, ps = (cnt >> 1) - 1;
    outside_level_kernel<<<cnt / 128, blk, 0, stream>>>(s2, s3, s0, W_out_inc, b_out_inc, s, cnt, ps);
  }
}
